// MaskGCNDisentangle_3204045603773
// MI455X (gfx1250) — matrix operand order repaired and checked by reading, not compiled, not run on hardware
//
#include <hip/hip_runtime.h>
#include <hip/hip_bf16.h>
#include <math.h>

// ---------------- problem constants ----------------
#define N_NODES 2048
#define N_EDGES 32768
#define EMBED   256
#define HIDDEN  128
#define BATCH   16
#define M_ROWS  (N_NODES * BATCH)        // 32768 flattened (node,batch) rows
#define NODE_FEATS (BATCH * EMBED)       // 4096 floats per node
#define OUT_HALF ((size_t)N_NODES * NODE_FEATS)

typedef __bf16 bf16_t;
typedef __attribute__((ext_vector_type(16))) __bf16 v16bf;
typedef __attribute__((ext_vector_type(8)))  __bf16 v8bf;
typedef __attribute__((ext_vector_type(8)))  float  v8f;

// ---------------- WMMA fragment loads (contiguous k = 16*half + s, same for A and B) ----
// A (16x32 bf16, row-major [rows x K]): lane m = lane&15, half = lane>>4.
// elements 0..7  = K in [k0 + half*16,     +8)
// elements 8..15 = K in [k0 + half*16 + 8, +8)
__device__ __forceinline__ v16bf load_a_frag(const bf16_t* __restrict__ A,
                                             int K, int row, int k0, int half) {
    const bf16_t* base = A + (size_t)row * K;
    const v8bf lo = *reinterpret_cast<const v8bf*>(base + k0 + half * 16);
    const v8bf hi = *reinterpret_cast<const v8bf*>(base + k0 + half * 16 + 8);
    v16bf a;
#pragma unroll
    for (int i = 0; i < 8; ++i) { a[i] = lo[i]; a[8 + i] = hi[i]; }
    return a;
}

// B (32x16 bf16), stored pre-transposed Bt[N][K]: lane n = lane&15,
// per-lane 16 contiguous K values at k0 + half*16.
__device__ __forceinline__ v16bf load_b_frag(const bf16_t* __restrict__ Bt,
                                             int K, int col, int k0, int half) {
    return *reinterpret_cast<const v16bf*>(Bt + (size_t)col * K + k0 + half * 16);
}

// Each wave computes a 16(M) x 64(N) C tile: one A fragment is reused by
// 4 WMMAs per K-step (4x less A-side L2 traffic, 4x higher wmma density).
template <int K>
__device__ __forceinline__ void gemm_tile_16x64(
        const bf16_t* __restrict__ A, const bf16_t* __restrict__ Bt,
        int row0, int col0, int half, int lid, v8f acc[4]) {
#pragma unroll
    for (int k0 = 0; k0 < K; k0 += 32) {
        const v16bf a = load_a_frag(A, K, row0 + lid, k0, half);
#pragma unroll
        for (int j = 0; j < 4; ++j) {
            const v16bf b = load_b_frag(Bt, K, col0 + j * 16 + lid, k0, half);
            acc[j] = __builtin_amdgcn_wmma_f32_16x16x32_bf16(
                false, a, false, b, (short)0, acc[j], false, false);
        }
    }
}

// ---------------- GEMM 1: Z1 = relu(X*W1 + b1), out bf16 ----------------
template <int K, int N>
__global__ __launch_bounds__(256) void gemm_bias_relu_kernel(
        const bf16_t* __restrict__ A, const bf16_t* __restrict__ Bt,
        const float* __restrict__ bias, bf16_t* __restrict__ Out) {
    const int lane = threadIdx.x & 31;
    const int gw   = (blockIdx.x * blockDim.x + threadIdx.x) >> 5;
    const int tm = gw / (N >> 6), tn = gw % (N >> 6);
    const int row0 = tm << 4, col0 = tn << 6;
    const int half = lane >> 4, lid = lane & 15;
    v8f acc[4] = {};
    gemm_tile_16x64<K>(A, Bt, row0, col0, half, lid, acc);
#pragma unroll
    for (int j = 0; j < 4; ++j) {
        const int n = col0 + j * 16 + lid;
        const float bn = bias[n];
#pragma unroll
        for (int r = 0; r < 8; ++r) {
            const int m = row0 + r + half * 8;
            float z = acc[j][r] + bn;
            z = z > 0.0f ? z : 0.0f;
            Out[(size_t)m * N + n] = (bf16_t)z;
        }
    }
}

// -------- GEMM 2: Z = Z1*W2 + b2; epilogue -> sigmoid masks * X -----------
template <int K, int N>
__global__ __launch_bounds__(256) void gemm_mask_kernel(
        const bf16_t* __restrict__ A, const bf16_t* __restrict__ Bt,
        const float* __restrict__ bias, const float* __restrict__ X,
        bf16_t* __restrict__ Xadj, bf16_t* __restrict__ Xconf) {
    const int lane = threadIdx.x & 31;
    const int gw   = (blockIdx.x * blockDim.x + threadIdx.x) >> 5;
    const int tm = gw / (N >> 6), tn = gw % (N >> 6);
    const int row0 = tm << 4, col0 = tn << 6;
    const int half = lane >> 4, lid = lane & 15;
    v8f acc[4] = {};
    gemm_tile_16x64<K>(A, Bt, row0, col0, half, lid, acc);
#pragma unroll
    for (int j = 0; j < 4; ++j) {
        const int n = col0 + j * 16 + lid;
        const float bn = bias[n];
#pragma unroll
        for (int r = 0; r < 8; ++r) {
            const int m = row0 + r + half * 8;
            const float z = acc[j][r] + bn;
            const float s = 1.0f / (1.0f + __expf(-z));   // sigmoid(z)
            const float xv = X[(size_t)m * N + n];
            Xadj[(size_t)m * N + n]  = (bf16_t)((1.0f - s) * xv); // sigmoid(-z)*x
            Xconf[(size_t)m * N + n] = (bf16_t)(s * xv);          // sigmoid(z)*x
        }
    }
}

// ---------------- GEMM 3/4: XW = Xmask * W, out bf16 ----------------------
template <int K, int N>
__global__ __launch_bounds__(256) void gemm_plain_kernel(
        const bf16_t* __restrict__ A, const bf16_t* __restrict__ Bt,
        bf16_t* __restrict__ Out) {
    const int lane = threadIdx.x & 31;
    const int gw   = (blockIdx.x * blockDim.x + threadIdx.x) >> 5;
    const int tm = gw / (N >> 6), tn = gw % (N >> 6);
    const int row0 = tm << 4, col0 = tn << 6;
    const int half = lane >> 4, lid = lane & 15;
    v8f acc[4] = {};
    gemm_tile_16x64<K>(A, Bt, row0, col0, half, lid, acc);
#pragma unroll
    for (int j = 0; j < 4; ++j) {
        const int n = col0 + j * 16 + lid;
#pragma unroll
        for (int r = 0; r < 8; ++r) {
            const int m = row0 + r + half * 8;
            Out[(size_t)m * N + n] = (bf16_t)acc[j][r];
        }
    }
}

// ---------------- prep kernels -------------------------------------------
__global__ void f32_to_bf16_kernel(const float* __restrict__ in,
                                   bf16_t* __restrict__ out, int n) {
    int i = blockIdx.x * blockDim.x + threadIdx.x;
    if (i < n) out[i] = (bf16_t)in[i];
}

// in: [R][C] row-major f32 -> out: [C][R] bf16 (transposed)
__global__ void transpose_bf16_kernel(const float* __restrict__ in,
                                      bf16_t* __restrict__ out, int R, int C) {
    int i = blockIdx.x * blockDim.x + threadIdx.x;
    if (i < R * C) {
        int r = i / C, c = i % C;
        out[(size_t)c * R + r] = (bf16_t)in[i];
    }
}

// ---------------- graph normalization ------------------------------------
__global__ void deg_init_kernel(float* deg) {
    int i = blockIdx.x * blockDim.x + threadIdx.x;
    if (i < N_NODES) deg[i] = 1.0f;                  // self-loop
}
__global__ void deg_count_kernel(const int* __restrict__ ei, float* deg) {
    int e = blockIdx.x * blockDim.x + threadIdx.x;
    if (e < N_EDGES) atomicAdd(&deg[ei[N_EDGES + e]], 1.0f);   // dst degree
}
__global__ void dinv_kernel(const float* __restrict__ deg, float* dinv) {
    int i = blockIdx.x * blockDim.x + threadIdx.x;
    if (i < N_NODES) dinv[i] = rsqrtf(deg[i]);
}

// out[n] = bias + dinv[n]^2 * XW[n]  (bias + self-loop term; fully writes d_out)
__global__ void out_init_kernel(const bf16_t* __restrict__ XWa,
                                const bf16_t* __restrict__ XWc,
                                const float* __restrict__ b_adj,
                                const float* __restrict__ b_conf,
                                const float* __restrict__ dinv,
                                float* __restrict__ out) {
    size_t i = (size_t)blockIdx.x * blockDim.x + threadIdx.x;
    if (i < OUT_HALF) {
        int n = (int)(i / NODE_FEATS);
        int f = (int)(i % EMBED);
        float sl = dinv[n] * dinv[n];
        out[i]            = b_adj[f]  + sl * (float)XWa[i];
        out[i + OUT_HALF] = b_conf[f] + sl * (float)XWc[i];
    }
}

// per-edge scatter: out[dst] += dinv[src]*dinv[dst] * XW[src]
// unsafeAtomicAdd forces the hardware global_atomic_add_f32 (no CAS loop);
// the output arrays (67 MB) are L2-resident on the 192 MB L2.
__global__ __launch_bounds__(256) void edge_scatter_kernel(
        const int* __restrict__ ei, const float* __restrict__ dinv,
        const bf16_t* __restrict__ XWa, const bf16_t* __restrict__ XWc,
        float* __restrict__ out) {
    const int e = blockIdx.x;
    const int s = ei[e];
    const int d = ei[N_EDGES + e];
    const float nm = dinv[s] * dinv[d];
    const size_t sbase = (size_t)s * NODE_FEATS;
    const size_t dbase = (size_t)d * NODE_FEATS;
    float* outc = out + OUT_HALF;
    for (int i = threadIdx.x; i < NODE_FEATS; i += 256) {
        unsafeAtomicAdd(&out[dbase + i],  nm * (float)XWa[sbase + i]);
        unsafeAtomicAdd(&outc[dbase + i], nm * (float)XWc[sbase + i]);
    }
}

// ---------------- launch -------------------------------------------------
extern "C" void kernel_launch(void* const* d_in, const int* in_sizes, int n_in,
                              void* d_out, int out_size, void* d_ws, size_t ws_size,
                              hipStream_t stream) {
    const float* x      = (const float*)d_in[0];
    const int*   ei     = (const int*)  d_in[1];
    const float* w1     = (const float*)d_in[2];
    const float* b1     = (const float*)d_in[3];
    const float* w2     = (const float*)d_in[4];
    const float* b2     = (const float*)d_in[5];
    const float* w_adj  = (const float*)d_in[6];
    const float* b_adj  = (const float*)d_in[7];
    const float* w_conf = (const float*)d_in[8];
    const float* b_conf = (const float*)d_in[9];
    float* out = (float*)d_out;

    // workspace carve-up (256B aligned), total ~93 MB
    char* ws = (char*)d_ws;
    size_t off = 0;
    auto alloc = [&](size_t bytes) -> void* {
        void* p = ws + off;
        off += (bytes + 255) & ~(size_t)255;
        return p;
    };
    bf16_t* Xbf    = (bf16_t*)alloc((size_t)M_ROWS * EMBED  * 2);
    bf16_t* Z1bf   = (bf16_t*)alloc((size_t)M_ROWS * HIDDEN * 2);
    bf16_t* Xadjbf = (bf16_t*)alloc((size_t)M_ROWS * EMBED  * 2);
    bf16_t* Xconfbf= (bf16_t*)alloc((size_t)M_ROWS * EMBED  * 2);
    bf16_t* XWa    = (bf16_t*)alloc((size_t)M_ROWS * EMBED  * 2);
    bf16_t* XWc    = (bf16_t*)alloc((size_t)M_ROWS * EMBED  * 2);
    bf16_t* w1t    = (bf16_t*)alloc((size_t)HIDDEN * EMBED  * 2); // [128][256]
    bf16_t* w2t    = (bf16_t*)alloc((size_t)EMBED  * HIDDEN * 2); // [256][128]
    bf16_t* wadjt  = (bf16_t*)alloc((size_t)EMBED  * EMBED  * 2); // [256][256]
    bf16_t* wconft = (bf16_t*)alloc((size_t)EMBED  * EMBED  * 2);
    float*  deg    = (float*) alloc((size_t)N_NODES * 4);
    float*  dinv   = (float*) alloc((size_t)N_NODES * 4);

    const int T = 256;
    // --- precision prep: bf16 copies / transposed weights ---
    {
        int n = M_ROWS * EMBED;
        f32_to_bf16_kernel<<<(n + T - 1) / T, T, 0, stream>>>(x, Xbf, n);
    }
    transpose_bf16_kernel<<<(EMBED * HIDDEN + T - 1) / T, T, 0, stream>>>(w1, w1t, EMBED, HIDDEN);
    transpose_bf16_kernel<<<(HIDDEN * EMBED + T - 1) / T, T, 0, stream>>>(w2, w2t, HIDDEN, EMBED);
    transpose_bf16_kernel<<<(EMBED * EMBED + T - 1) / T, T, 0, stream>>>(w_adj, wadjt, EMBED, EMBED);
    transpose_bf16_kernel<<<(EMBED * EMBED + T - 1) / T, T, 0, stream>>>(w_conf, wconft, EMBED, EMBED);

    // --- GCN symmetric normalization ---
    deg_init_kernel<<<(N_NODES + T - 1) / T, T, 0, stream>>>(deg);
    deg_count_kernel<<<(N_EDGES + T - 1) / T, T, 0, stream>>>(ei, deg);
    dinv_kernel<<<(N_NODES + T - 1) / T, T, 0, stream>>>(deg, dinv);

    // --- mask MLP (two WMMA GEMMs); waves = (M/16)*(N/64), 8 waves/block ---
    // Z1 = relu(X*W1+b1): (32768/16)*(128/64) = 4096 waves -> 512 blocks
    gemm_bias_relu_kernel<EMBED, HIDDEN><<<4096 / 8, T, 0, stream>>>(Xbf, w1t, b1, Z1bf);
    // Z = Z1*W2+b2 -> masks: 2048*4 = 8192 waves -> 1024 blocks
    gemm_mask_kernel<HIDDEN, EMBED><<<8192 / 8, T, 0, stream>>>(Z1bf, w2t, b2, x, Xadjbf, Xconfbf);

    // --- per-node feature transforms (two WMMA GEMMs) ---
    gemm_plain_kernel<EMBED, EMBED><<<8192 / 8, T, 0, stream>>>(Xadjbf,  wadjt,  XWa);
    gemm_plain_kernel<EMBED, EMBED><<<8192 / 8, T, 0, stream>>>(Xconfbf, wconft, XWc);

    // --- aggregate: bias + self-loop, then per-edge scatter-add ---
    out_init_kernel<<<(int)((OUT_HALF + T - 1) / T), T, 0, stream>>>(XWa, XWc, b_adj, b_conf, dinv, out);
    edge_scatter_kernel<<<N_EDGES, T, 0, stream>>>(ei, dinv, XWa, XWc, out);
}
